// MultiHeadAttention3D_20280835572057
// MI455X (gfx1250) — matrix operand order repaired and checked by reading, not compiled, not run on hardware
//
#include <hip/hip_runtime.h>
#include <hip/hip_bf16.h>

#define BB   2
#define CH   256
#define NH   8
#define HD   32
#define NVOX 4096

typedef _Float16 h2_t  __attribute__((ext_vector_type(2)));
typedef _Float16 h8_t  __attribute__((ext_vector_type(8)));
typedef _Float16 h16_t __attribute__((ext_vector_type(16)));
typedef float    f4_t  __attribute__((ext_vector_type(4)));
typedef float    v8f   __attribute__((ext_vector_type(8)));

static __device__ __forceinline__ h2_t pkrtz(float a, float b) {
  return __builtin_bit_cast(h2_t, __builtin_amdgcn_cvt_pkrtz(a, b));
}

static __device__ __forceinline__ h16_t hcat(h8_t a, h8_t b) {
  return __builtin_shufflevector(a, b, 0,1,2,3,4,5,6,7,8,9,10,11,12,13,14,15);
}

static __device__ __forceinline__ v8f wmma16(h16_t a, h16_t b, v8f c) {
  // D = A(16x32 f16) * B(32x16 f16) + C(16x16 f32)
  return __builtin_amdgcn_wmma_f32_16x16x32_f16(false, a, false, b, (short)0, c, false, false);
}

// Pack 8 f32 (two float4) into f16 lanes [base..base+7] of A.
static __device__ __forceinline__ void pack8(h16_t& A, int base, f4_t u, f4_t v) {
  h2_t t0 = pkrtz(u[0], u[1]);
  h2_t t1 = pkrtz(u[2], u[3]);
  h2_t t2 = pkrtz(v[0], v[1]);
  h2_t t3 = pkrtz(v[2], v[3]);
  A[base+0]=t0[0]; A[base+1]=t0[1]; A[base+2]=t1[0]; A[base+3]=t1[1];
  A[base+4]=t2[0]; A[base+5]=t2[1]; A[base+6]=t3[0]; A[base+7]=t3[1];
}

// Async DMA of one 16B chunk from global into LDS (ASYNCcnt-tracked).
static __device__ __forceinline__ void async_b128(unsigned lds_addr, const void* gsrc) {
  asm volatile("global_load_async_to_lds_b128 %0, %1, off"
               :: "v"(lds_addr), "v"(gsrc) : "memory");
}

// ---------------------------------------------------------------------------
// Stage 1: QKV = qkv_w[768x256] @ x[256x4096] per batch (f32 in, f16 out)
// Q,K stored [b][h][n][hd]; V stored [b][h][hd][n]. Q pre-scaled by hd^-0.5.
// ---------------------------------------------------------------------------
__global__ __launch_bounds__(256) void mha3d_qkv_kernel(
    const float* __restrict__ x, const float* __restrict__ w,
    _Float16* __restrict__ qws, _Float16* __restrict__ kws,
    _Float16* __restrict__ vws) {
  const int lane = threadIdx.x & 31;
  const int wv   = threadIdx.x >> 5;
  const int hi   = lane >> 4;
  const int ln   = lane & 15;
  const int gtile = blockIdx.x * 8 + wv;          // 0 .. 24575
  const int b   = gtile / (48*256);
  const int rem = gtile % (48*256);
  const int rt  = rem >> 8;                       // 0..47 (out-channel tile)
  const int ct  = rem & 255;                      // 0..255 (voxel tile)
  const int orow0 = rt * 16;
  const int col0  = ct * 16;

  v8f acc = {};
  const float* wrow = w + (size_t)(orow0 + ln) * CH;
  const float* xcol = x + (size_t)b * CH * NVOX + (col0 + ln);

  #pragma unroll
  for (int kk = 0; kk < 8; ++kk) {
    const int k0 = kk * 32;
    const float* wr = wrow + k0 + hi * 8;
    h16_t A;
    pack8(A, 0, *(const f4_t*)(wr),      *(const f4_t*)(wr + 4));
    pack8(A, 8, *(const f4_t*)(wr + 16), *(const f4_t*)(wr + 20));
    h16_t Bm;
    #pragma unroll
    for (int j = 0; j < 8; ++j) {
      const float* xp = xcol + (size_t)(k0 + hi * 8 + (j >> 2) * 16 + 2 * (j & 3)) * NVOX;
      h2_t t = pkrtz(xp[0], xp[NVOX]);
      Bm[2*j] = t[0]; Bm[2*j+1] = t[1];
    }
    acc = wmma16(A, Bm, acc);
  }

  const int n = col0 + ln;
  if (rt < 16) {                       // Q : scale and store [n][hd]
    const int ch0 = orow0 + hi * 8;
    const int hh = ch0 >> 5, d0 = ch0 & 31;
    _Float16* dst = qws + ((size_t)(b*NH + hh) * NVOX + n) * HD + d0;
    h8_t o;
    #pragma unroll
    for (int r = 0; r < 8; ++r) o[r] = (_Float16)(acc[r] * 0.17677669529663687f);
    *(h8_t*)dst = o;
  } else if (rt < 32) {                // K : store [n][hd]
    const int ch0 = orow0 - 256 + hi * 8;
    const int hh = ch0 >> 5, d0 = ch0 & 31;
    _Float16* dst = kws + ((size_t)(b*NH + hh) * NVOX + n) * HD + d0;
    h8_t o;
    #pragma unroll
    for (int r = 0; r < 8; ++r) o[r] = (_Float16)acc[r];
    *(h8_t*)dst = o;
  } else {                             // V : store [hd][n]
    const int ch0 = orow0 - 512 + hi * 8;
    const int hh = ch0 >> 5, d0 = ch0 & 31;
    _Float16* dst = vws + ((size_t)(b*NH + hh) * HD + d0) * NVOX + n;
    #pragma unroll
    for (int r = 0; r < 8; ++r) dst[(size_t)r * NVOX] = (_Float16)acc[r];
  }
}

// ---------------------------------------------------------------------------
// Stage 2: flash attention per (b, head). Grid = B*NH*(N/128) = 512 blocks,
// 8 waves/block, each wave owns 16 query rows. K/V tiles are staged into LDS
// once per block with async DMA (double-buffered), so operands come from
// ds_load_b128 instead of 8x-redundant global reads. Computes S^T = K*Q^T so
// softmaxed P is already in WMMA A-layout (lane-local f32->f16 pack only).
// ---------------------------------------------------------------------------
__global__ __launch_bounds__(256) void mha3d_attn_kernel(
    const _Float16* __restrict__ qws, const _Float16* __restrict__ kws,
    const _Float16* __restrict__ vws, _Float16* __restrict__ aows) {
  __shared__ __align__(16) _Float16 ksh[2][128 * HD];  // [key][hd], 8KB each
  __shared__ __align__(16) _Float16 vsh[2][HD * 128];  // [hd][key], 8KB each

  const int tid  = threadIdx.x;
  const int lane = tid & 31;
  const int wv   = tid >> 5;
  const int hi   = lane >> 4;
  const int ln   = lane & 15;
  const int qt   = blockIdx.x & 31;     // query tile (128 queries)
  const int bh   = blockIdx.x >> 5;     // 0..15 = b*NH + h
  const int q0   = qt * 128 + wv * 16;

  const _Float16* qp = qws + (size_t)bh * NVOX * HD;
  const _Float16* kp = kws + (size_t)bh * NVOX * HD;
  const _Float16* vp = vws + (size_t)bh * HD * NVOX;
  const int bb = bh >> 3, hh = bh & 7;
  _Float16* aop = aows + (size_t)bb * NVOX * CH + hh * HD;

  // Q as B-operand of S^T = K*Q^T : lane = hi*16 + q, d = 8*hi + s (s<8), 16 + 8*hi + (s-8) (s>=8)
  const _Float16* qptr = qp + (size_t)(q0 + ln) * HD + hi * 8;
  const h16_t Qb = hcat(*(const h8_t*)qptr, *(const h8_t*)(qptr + 16));

  // Issue async stage of K/V tile `kb` (128 keys) into buffer `buf`.
  auto stage = [&](int kbase, int buf) {
    const char* kg = (const char*)(kp + (size_t)kbase * HD);   // [key][hd] rows, 64B
    const unsigned kl = (unsigned)(uintptr_t)&ksh[buf][0];
    #pragma unroll
    for (int i = 0; i < 2; ++i) {
      const int c = tid + i * 256;                 // 16B chunk id, 0..511
      async_b128(kl + c * 16, kg + c * 16);
    }
    const unsigned vl = (unsigned)(uintptr_t)&vsh[buf][0];
    #pragma unroll
    for (int i = 0; i < 2; ++i) {
      const int c = tid + i * 256;
      const int d = c >> 4, o = c & 15;            // row d (256B), chunk o
      const char* vg = (const char*)(vp + (size_t)d * NVOX + kbase);
      async_b128(vl + c * 16, vg + o * 16);
    }
  };

  v8f o0 = {}, o1 = {};                 // O[q][d] accumulators, d tiles 0/1
  float mprev = -3.0e38f, lsum = 0.0f;

  stage(0, 0);

  for (int kt = 0; kt < 32; ++kt) {
    const int cur = kt & 1;
    if (kt + 1 < 32) {
      stage((kt + 1) * 128, cur ^ 1);
      asm volatile("s_wait_asynccnt 0x4" ::: "memory");  // tile kt's 4 chunks done
    } else {
      asm volatile("s_wait_asynccnt 0x0" ::: "memory");
    }
    __syncthreads();                    // whole tile visible to all waves

    float s[8][8];                      // S^T tiles: key = 16t+8hi+r, q = q0+ln
    #pragma unroll
    for (int j = 0; j < 8; ++j) {
      const _Float16* kr = &ksh[cur][(j * 16 + ln) * HD + hi * 8];
      h16_t A = hcat(*(const h8_t*)kr, *(const h8_t*)(kr + 16));
      v8f c = {};
      c = wmma16(A, Qb, c);
      #pragma unroll
      for (int r = 0; r < 8; ++r) s[j][r] = c[r];
    }

    // online softmax over keys (per-lane scalar stats for q = ln)
    float m = s[0][0];
    #pragma unroll
    for (int j = 0; j < 8; ++j)
      #pragma unroll
      for (int r = 0; r < 8; ++r) m = fmaxf(m, s[j][r]);
    m = fmaxf(m, __shfl_xor(m, 16, 32));
    const float mnew  = fmaxf(mprev, m);
    const float alpha = __expf(mprev - mnew);
    float rs = 0.0f;
    #pragma unroll
    for (int j = 0; j < 8; ++j)
      #pragma unroll
      for (int r = 0; r < 8; ++r) {
        float p = __expf(s[j][r] - mnew);
        s[j][r] = p;
        rs += p;
      }
    rs += __shfl_xor(rs, 16, 32);
    lsum = lsum * alpha + rs;
    mprev = mnew;

    // rescale O rows (row q = q0 + r + 8*hi -> alpha lives at lane r+8*hi)
    #pragma unroll
    for (int r = 0; r < 8; ++r) {
      const float al = __shfl(alpha, r + hi * 8, 32);
      o0[r] *= al; o1[r] *= al;
    }

    // O += P * V over four 32-key chunks
    #pragma unroll
    for (int c4 = 0; c4 < 4; ++c4) {
      h16_t P;
      #pragma unroll
      for (int j = 0; j < 4; ++j) {
        h2_t t = pkrtz(s[2*c4][2*j],   s[2*c4][2*j+1]);
        P[2*j]   = t[0]; P[2*j+1]   = t[1];
        h2_t u = pkrtz(s[2*c4+1][2*j], s[2*c4+1][2*j+1]);
        P[8+2*j] = u[0]; P[8+2*j+1] = u[1];
      }
      // V as B-operand: lane = hi*16 + d, key = 8*hi + s (s<8), 16 + 8*hi + (s-8) (s>=8); vsh is [d][key]
      const _Float16* v0 = &vsh[cur][ln * 128        + c4 * 32 + hi * 8];
      const _Float16* v1 = &vsh[cur][(16 + ln) * 128 + c4 * 32 + hi * 8];
      h16_t Bv0 = hcat(*(const h8_t*)v0, *(const h8_t*)(v0 + 16));
      o0 = wmma16(P, Bv0, o0);
      h16_t Bv1 = hcat(*(const h8_t*)v1, *(const h8_t*)(v1 + 16));
      o1 = wmma16(P, Bv1, o1);
    }

    __syncthreads();                    // all reads done before buffer reuse
  }

  const float linv = 1.0f / lsum;
  #pragma unroll
  for (int r = 0; r < 8; ++r) {
    const float sc = __shfl(linv, r + hi * 8, 32);
    const int qrow = q0 + r + hi * 8;
    aop[(size_t)qrow * CH + ln]      = (_Float16)(o0[r] * sc);
    aop[(size_t)qrow * CH + 16 + ln] = (_Float16)(o1[r] * sc);
  }
}

// ---------------------------------------------------------------------------
// Stage 3: OUT = proj_w[256x256] @ AO[256x4096] + bias, f32 output.
// AO is stored [b][n][c] f16, so B-operand loads are contiguous b128.
// ---------------------------------------------------------------------------
__global__ __launch_bounds__(256) void mha3d_proj_kernel(
    const _Float16* __restrict__ ao, const float* __restrict__ w,
    const float* __restrict__ bias, float* __restrict__ out) {
  const int lane = threadIdx.x & 31;
  const int wv   = threadIdx.x >> 5;
  const int hi   = lane >> 4;
  const int ln   = lane & 15;
  const int gtile = blockIdx.x * 8 + wv;         // 0 .. 8191
  const int b   = gtile / (16*256);
  const int rem = gtile % (16*256);
  const int rt  = rem >> 8;                      // 0..15
  const int ct  = rem & 255;                     // 0..255
  const int orow0 = rt * 16;
  const int col0  = ct * 16;

  v8f acc = {};
  const float*    wrow = w  + (size_t)(orow0 + ln) * CH;
  const _Float16* bcol = ao + (size_t)b * NVOX * CH + (size_t)(col0 + ln) * CH;

  #pragma unroll
  for (int kk = 0; kk < 8; ++kk) {
    const int k0 = kk * 32;
    const float* wr = wrow + k0 + hi * 8;
    h16_t A;
    pack8(A, 0, *(const f4_t*)(wr),      *(const f4_t*)(wr + 4));
    pack8(A, 8, *(const f4_t*)(wr + 16), *(const f4_t*)(wr + 20));
    const _Float16* bp = bcol + k0 + hi * 8;
    h16_t Bm = hcat(*(const h8_t*)bp, *(const h8_t*)(bp + 16));
    acc = wmma16(A, Bm, acc);
  }

  const int n = col0 + ln;
  #pragma unroll
  for (int r = 0; r < 8; ++r) {
    const int oc = orow0 + hi * 8 + r;
    out[((size_t)b * CH + oc) * NVOX + n] = acc[r] + bias[oc];
  }
}

// ---------------------------------------------------------------------------
extern "C" void kernel_launch(void* const* d_in, const int* in_sizes, int n_in,
                              void* d_out, int out_size, void* d_ws, size_t ws_size,
                              hipStream_t stream) {
  const float* x      = (const float*)d_in[0];
  const float* qkv_w  = (const float*)d_in[1];
  const float* proj_w = (const float*)d_in[2];
  const float* proj_b = (const float*)d_in[3];
  float* out = (float*)d_out;

  const size_t per = (size_t)BB * NH * NVOX * HD;  // 2M f16 elements each
  _Float16* qws  = (_Float16*)d_ws;
  _Float16* kws  = qws + per;
  _Float16* vws  = kws + per;
  _Float16* aows = vws + per;                      // [b][n][c], 2M f16

  mha3d_qkv_kernel <<<3072, 256, 0, stream>>>(x, qkv_w, qws, kws, vws);
  mha3d_attn_kernel<<<512,  256, 0, stream>>>(qws, kws, vws, aows);
  mha3d_proj_kernel<<<1024, 256, 0, stream>>>(aows, proj_w, proj_b, out);
}
